// LSTM_13692355740404
// MI455X (gfx1250) — matrix operand order repaired and checked by reading, not compiled, not run on hardware
//
#include <hip/hip_runtime.h>
#include <hip/hip_bf16.h>

typedef __bf16 v16bf  __attribute__((ext_vector_type(16)));
typedef __bf16 bf16x8 __attribute__((ext_vector_type(8)));
typedef __bf16 bf16x4 __attribute__((ext_vector_type(4)));
typedef float  v8f    __attribute__((ext_vector_type(8)));
typedef float  f32x4  __attribute__((ext_vector_type(4)));

#define B_   128
#define T_   512
#define D_   512
#define H_   1024
#define G4_  4096   // 4*H

// scheduling-group masks (LLVM AMDGPU sched_group_barrier)
#define SG_WMMA      0x008
#define SG_VMEM_READ 0x020

// ---- d_ws layout (bytes) ----
// 0        : h buffer 0, bf16 [B,H]  (262144)
// 262144   : h buffer 1, bf16 [B,H]  (262144)
// 524288   : c state,    f32 [B,H]   (524288)
// 1048576  : W_ih bf16 [4096,512]    (4194304)
// 5242880  : W_hh bf16 [4096,1024]   (8388608)
// total    : 13631488 (~13 MB)

__global__ __launch_bounds__(256) void cvt_weights_bf16(
    const float* __restrict__ wih, const float* __restrict__ whh,
    __bf16* __restrict__ wihb, __bf16* __restrict__ whhb)
{
    const int NIH4 = (G4_ * D_) / 4;   // 524288 float4s
    const int NHH4 = (G4_ * H_) / 4;   // 1048576 float4s
    int i = blockIdx.x * blockDim.x + threadIdx.x;
    if (i >= NIH4 + NHH4) return;
    if (i < NIH4) {
        f32x4 v = ((const f32x4*)wih)[i];
        bf16x4 o;
        o[0] = (__bf16)v[0]; o[1] = (__bf16)v[1];
        o[2] = (__bf16)v[2]; o[3] = (__bf16)v[3];
        ((bf16x4*)wihb)[i] = o;
    } else {
        int j = i - NIH4;
        f32x4 v = ((const f32x4*)whh)[j];
        bf16x4 o;
        o[0] = (__bf16)v[0]; o[1] = (__bf16)v[1];
        o[2] = (__bf16)v[2]; o[3] = (__bf16)v[3];
        ((bf16x4*)whhb)[j] = o;
    }
}

// A fragment from f32 row (16-bit A 16x32 layout), cvt in-register.
__device__ __forceinline__ v16bf load_a_f32(const float* __restrict__ p)
{
    f32x4 a0 = *(const f32x4*)(p);
    f32x4 a1 = *(const f32x4*)(p + 4);
    f32x4 a2 = *(const f32x4*)(p + 8);
    f32x4 a3 = *(const f32x4*)(p + 12);
    v16bf a;
#pragma unroll
    for (int e = 0; e < 4; ++e) {
        a[e]      = (__bf16)a0[e];
        a[4 + e]  = (__bf16)a1[e];
        a[8 + e]  = (__bf16)a2[e];
        a[12 + e] = (__bf16)a3[e];
    }
    return a;
}

// A fragment from bf16 row (no conversion needed).
__device__ __forceinline__ v16bf load_a_bf16(const __bf16* __restrict__ p)
{
    bf16x8 lo = *(const bf16x8*)(p);
    bf16x8 hi = *(const bf16x8*)(p + 8);
    v16bf a;
#pragma unroll
    for (int e = 0; e < 8; ++e) { a[e] = lo[e]; a[8 + e] = hi[e]; }
    return a;
}

#define WMMA_BF16(A, Bv, C) \
    __builtin_amdgcn_wmma_f32_16x16x32_bf16(false, (A), false, (Bv), (short)0, (C), false, false)

// x-projection K-loop (f32 A), distance-1 software pipeline enforced with
// sched_group_barrier: per iteration, 12 VMEM reads (next A+B) must precede
// the 4 WMMAs consuming the current fragments.
__device__ __forceinline__ void gemm_x(
    const float* __restrict__ arow, const __bf16* __restrict__ brow,
    v8f acc[4])
{
    const size_t gs = (size_t)H_ * D_;
    v16bf aC = load_a_f32(arow);
    v16bf b0 = *(const v16bf*)(brow);
    v16bf b1 = *(const v16bf*)(brow + gs);
    v16bf b2 = *(const v16bf*)(brow + 2 * gs);
    v16bf b3 = *(const v16bf*)(brow + 3 * gs);
#pragma unroll 2
    for (int k0 = 32; k0 < D_; k0 += 32) {
        v16bf aN = load_a_f32(arow + k0);
        v16bf n0 = *(const v16bf*)(brow + k0);
        v16bf n1 = *(const v16bf*)(brow + gs + k0);
        v16bf n2 = *(const v16bf*)(brow + 2 * gs + k0);
        v16bf n3 = *(const v16bf*)(brow + 3 * gs + k0);
        acc[0] = WMMA_BF16(aC, b0, acc[0]);
        acc[1] = WMMA_BF16(aC, b1, acc[1]);
        acc[2] = WMMA_BF16(aC, b2, acc[2]);
        acc[3] = WMMA_BF16(aC, b3, acc[3]);
        __builtin_amdgcn_sched_group_barrier(SG_VMEM_READ, 12, 0);
        __builtin_amdgcn_sched_group_barrier(SG_WMMA, 4, 0);
        aC = aN; b0 = n0; b1 = n1; b2 = n2; b3 = n3;
    }
    acc[0] = WMMA_BF16(aC, b0, acc[0]);
    acc[1] = WMMA_BF16(aC, b1, acc[1]);
    acc[2] = WMMA_BF16(aC, b2, acc[2]);
    acc[3] = WMMA_BF16(aC, b3, acc[3]);
}

// h-projection K-loop (bf16 A): 10 VMEM reads per iteration before 4 WMMAs.
__device__ __forceinline__ void gemm_h(
    const __bf16* __restrict__ arow, const __bf16* __restrict__ brow,
    v8f acc[4])
{
    const size_t gs = (size_t)H_ * H_;
    v16bf aC = load_a_bf16(arow);
    v16bf b0 = *(const v16bf*)(brow);
    v16bf b1 = *(const v16bf*)(brow + gs);
    v16bf b2 = *(const v16bf*)(brow + 2 * gs);
    v16bf b3 = *(const v16bf*)(brow + 3 * gs);
#pragma unroll 2
    for (int k0 = 32; k0 < H_; k0 += 32) {
        v16bf aN = load_a_bf16(arow + k0);
        v16bf n0 = *(const v16bf*)(brow + k0);
        v16bf n1 = *(const v16bf*)(brow + gs + k0);
        v16bf n2 = *(const v16bf*)(brow + 2 * gs + k0);
        v16bf n3 = *(const v16bf*)(brow + 3 * gs + k0);
        acc[0] = WMMA_BF16(aC, b0, acc[0]);
        acc[1] = WMMA_BF16(aC, b1, acc[1]);
        acc[2] = WMMA_BF16(aC, b2, acc[2]);
        acc[3] = WMMA_BF16(aC, b3, acc[3]);
        __builtin_amdgcn_sched_group_barrier(SG_VMEM_READ, 10, 0);
        __builtin_amdgcn_sched_group_barrier(SG_WMMA, 4, 0);
        aC = aN; b0 = n0; b1 = n1; b2 = n2; b3 = n3;
    }
    acc[0] = WMMA_BF16(aC, b0, acc[0]);
    acc[1] = WMMA_BF16(aC, b1, acc[1]);
    acc[2] = WMMA_BF16(aC, b2, acc[2]);
    acc[3] = WMMA_BF16(aC, b3, acc[3]);
}

// One LSTM timestep: gates = x_t @ W_ih^T + h_in @ W_hh^T + b ; update (h,c).
// Wave tile: 16 batch rows x 16 hidden cols, all 4 gates in one wave so the
// nonlinearity fuses in-register with the WMMA accumulators.
__global__ __launch_bounds__(256) void lstm_step(
    const float*  __restrict__ x,     // [B,T,D] f32
    const __bf16* __restrict__ wih,   // [4096,512] bf16
    const __bf16* __restrict__ whh,   // [4096,1024] bf16
    const float*  __restrict__ b_ih,  // [4096]
    const float*  __restrict__ b_hh,  // [4096]
    const __bf16* __restrict__ h_in,  // [B,H] bf16
    __bf16*       __restrict__ h_out, // [B,H] bf16
    float*        __restrict__ c_st,  // [B,H] f32, in/out
    float*        __restrict__ fout,  // final-step f32 output or nullptr
    int t)
{
    const int lane = threadIdx.x & 31;
    const int wid  = (blockIdx.x << 3) + (threadIdx.x >> 5); // 0..511
    const int mt   = wid & 7;      // batch tile (8x16); all 8 waves of a block
    const int nh   = wid >> 3;     // share nh -> identical weight rows (L0 hit)
    const int m0   = mt * 16;
    const int n0   = nh * 16;
    const int hl   = lane >> 4;    // half-wave selector
    const int nl   = lane & 15;

    // A fragment (16-bit A 16x32): lanes 0-15 -> K 0..15,
    // lanes 16-31 -> K 16..31.  B fragment (32x16): column = nl,
    // K base = hl*16, 16 contiguous k per lane.
    const int am   = m0 + nl;
    const int akb  = hl * 16;
    const int bkb  = hl * 16;

    v8f acc[4] = {};               // gates i,f,g,o

    gemm_x(x + (size_t)am * (T_ * D_) + (size_t)t * D_ + akb,
           wih + (size_t)(n0 + nl) * D_ + bkb, acc);
    gemm_h(h_in + (size_t)am * H_ + akb,
           whh + (size_t)(n0 + nl) * H_ + bkb, acc);

    // epilogue: C/D layout -> lane column n = n0+nl ; VGPR j -> row m0+j
    // (+8 for upper half-wave).
    const int n  = n0 + nl;
    const float bi = b_ih[n]          + b_hh[n];
    const float bf = b_ih[H_ + n]     + b_hh[H_ + n];
    const float bg = b_ih[2 * H_ + n] + b_hh[2 * H_ + n];
    const float bo = b_ih[3 * H_ + n] + b_hh[3 * H_ + n];
#pragma unroll
    for (int j = 0; j < 8; ++j) {
        const int m = m0 + j + hl * 8;
        const size_t idx = (size_t)m * H_ + n;
        float iv = acc[0][j] + bi;
        float fv = acc[1][j] + bf;
        float gv = acc[2][j] + bg;
        float ov = acc[3][j] + bo;
        float ig = 1.0f / (1.0f + __expf(-iv));
        float fg = 1.0f / (1.0f + __expf(-fv));
        float og = 1.0f / (1.0f + __expf(-ov));
        float eg = __expf(-2.0f * gv);
        float gg = (1.0f - eg) / (1.0f + eg);        // tanh(g)
        float cn = fg * c_st[idx] + ig * gg;
        float ec = __expf(-2.0f * cn);
        float tc = (1.0f - ec) / (1.0f + ec);        // tanh(c)
        float hv = og * tc;
        c_st[idx]  = cn;
        h_out[idx] = (__bf16)hv;
        if (fout) fout[idx] = hv;                    // final step: f32 output
    }
}

extern "C" void kernel_launch(void* const* d_in, const int* in_sizes, int n_in,
                              void* d_out, int out_size, void* d_ws, size_t ws_size,
                              hipStream_t stream) {
    (void)in_sizes; (void)n_in; (void)out_size; (void)ws_size;
    // Reference output is h0T: final hidden of LAYER 0 only. Layer-1 inputs
    // (d_in[5..8]) are dead w.r.t. the output.
    const float* x   = (const float*)d_in[0];
    const float* wih = (const float*)d_in[1];
    const float* whh = (const float*)d_in[2];
    const float* bih = (const float*)d_in[3];
    const float* bhh = (const float*)d_in[4];

    char* ws = (char*)d_ws;
    __bf16* hb0  = (__bf16*)(ws);
    __bf16* hb1  = (__bf16*)(ws + 262144);
    float*  c    = (float*)(ws + 524288);
    __bf16* wihb = (__bf16*)(ws + 1048576);
    __bf16* whhb = (__bf16*)(ws + 5242880);

    // zero h0, h1, c (graph-capture-safe async memset)
    hipMemsetAsync(ws, 0, 1048576, stream);

    // per-launch f32 -> bf16 weight conversion; stays L2-resident (12 MB).
    cvt_weights_bf16<<<6144, 256, 0, stream>>>(wih, whh, wihb, whhb);

    __bf16* hb[2] = { hb0, hb1 };
    for (int t = 0; t < T_; ++t) {
        const __bf16* hin  = hb[t & 1];
        __bf16*       hout = hb[(t + 1) & 1];
        float*        fo   = (t == T_ - 1) ? (float*)d_out : nullptr;
        lstm_step<<<64, 256, 0, stream>>>(x, wihb, whhb, bih, bhh,
                                          hin, hout, c, fo, t);
    }
}
